// LlamaAttention_70746701299832
// MI455X (gfx1250) — matrix operand order repaired and checked by reading, not compiled, not run on hardware
//
#include <hip/hip_runtime.h>
#include <math.h>

#define Bn 8
#define Qn 4
#define Hn 4096
#define NHn 32
#define NKVn 8
#define HDn 128
#define Gn 4
#define CAPn 4100
#define NSPLIT 8
#define SCALE_F 0.08838834764831845f  // 1/sqrt(128)

typedef __attribute__((ext_vector_type(16))) __bf16 v16bf;
typedef __attribute__((ext_vector_type(8)))  float v8f;

static __device__ inline v8f wmma_bf16(v16bf a, v16bf b, v8f c) {
  // D = A(16x32 bf16) * B(32x16 bf16) + C(16x16 f32)
  return __builtin_amdgcn_wmma_f32_16x16x32_bf16(false, a, false, b, (short)0, c,
                                                 false, false);
}

// gfx1250 async global->LDS copy (ASYNCcnt-tracked), per-lane 16B.
static __device__ inline void async_ld_b128(unsigned lds_off, const void* gptr) {
  asm volatile("global_load_async_to_lds_b128 %0, %1, off"
               :: "v"(lds_off), "v"((unsigned long long)(uintptr_t)gptr)
               : "memory");
}
static __device__ inline void wait_async_le0()  { asm volatile("s_wait_asynccnt 0x0"  ::: "memory"); }
static __device__ inline void wait_async_le4()  { asm volatile("s_wait_asynccnt 0x4"  ::: "memory"); }
static __device__ inline void wait_async_le8()  { asm volatile("s_wait_asynccnt 0x8"  ::: "memory"); }
static __device__ inline void wait_async_le12() { asm volatile("s_wait_asynccnt 0xc"  ::: "memory"); }
static __device__ inline void wait_async_le32() { asm volatile("s_wait_asynccnt 0x20" ::: "memory"); }

// ---------------------------------------------------------------------------
// Kernel 1: QKV projection. A = hidden (32 x 4096), B = [Wq|Wk|Wv] columns.
// One wave computes a 32x16 tile. Weight 32x16 k-tiles flow through a 4-deep
// async-LDS ring (coalesced b128 rows). Peeled pipeline: main loop has a
// single unconditional s_wait_asynccnt 12; 3-step epilogue drains the ring.
// ---------------------------------------------------------------------------
__global__ __launch_bounds__(32) void qkv_gemm_kernel(
    const float* __restrict__ hs, const float* __restrict__ Wq,
    const float* __restrict__ Wk, const float* __restrict__ Wv,
    float* __restrict__ qbuf, float* __restrict__ kbuf, float* __restrict__ vbuf) {
  __shared__ float wtile[4][32 * 16];

  const int lane = threadIdx.x & 31;
  const int half = lane >> 4, nr = lane & 15;
  const int cbase = blockIdx.x * 16;

  const float* W; int ncols; float* outp; int cb;
  if (cbase < NHn * HDn)               { W = Wq; ncols = NHn * HDn;  cb = cbase;                      outp = qbuf; }
  else if (cbase < (NHn + NKVn) * HDn) { W = Wk; ncols = NKVn * HDn; cb = cbase - NHn * HDn;          outp = kbuf; }
  else                                 { W = Wv; ncols = NKVn * HDn; cb = cbase - (NHn + NKVn) * HDn; outp = vbuf; }

  // per-lane staging pointers: 4 b128 slots covering 32 rows x 16 cols
  const float* gsrc[4];
  unsigned lbase[4];
  #pragma unroll
  for (int i = 0; i < 4; ++i) {
    const int slot = i * 32 + lane, row = slot >> 2, c4 = slot & 3;
    gsrc[i]  = W + (size_t)row * ncols + cb + c4 * 4;
    lbase[i] = (unsigned)(uintptr_t)&wtile[0][row * 16 + c4 * 4];
  }
  const size_t kadv = (size_t)32 * ncols;
  auto stage = [&](int buf) {
    #pragma unroll
    for (int i = 0; i < 4; ++i) {
      async_ld_b128(lbase[i] + (unsigned)buf * 2048u, gsrc[i]);
      gsrc[i] += kadv;
    }
  };

  v8f c0, c1;
  #pragma unroll
  for (int e = 0; e < 8; ++e) { c0[e] = 0.f; c1[e] = 0.f; }

  const float* h0 = hs + (size_t)nr * Hn + half * 8;
  const float* h1 = hs + (size_t)(16 + nr) * Hn + half * 8;

  auto body = [&](int cur) {
    // gather B column (batched DS loads), then convert
    float w[16];
    #pragma unroll
    for (int i = 0; i < 16; ++i) w[i] = wtile[cur][(i < 8 ? half * 8 + i : 8 + half * 8 + i) * 16 + nr];
    // A rows (coalesced b128), then convert
    float t0[16], t1[16];
    *(float4*)(t0 + 0)  = *(const float4*)(h0);
    *(float4*)(t0 + 4)  = *(const float4*)(h0 + 4);
    *(float4*)(t0 + 8)  = *(const float4*)(h0 + 16);
    *(float4*)(t0 + 12) = *(const float4*)(h0 + 20);
    *(float4*)(t1 + 0)  = *(const float4*)(h1);
    *(float4*)(t1 + 4)  = *(const float4*)(h1 + 4);
    *(float4*)(t1 + 8)  = *(const float4*)(h1 + 16);
    *(float4*)(t1 + 12) = *(const float4*)(h1 + 20);
    v16bf bm, a0, a1;
    #pragma unroll
    for (int i = 0; i < 16; ++i) {
      bm[i] = (__bf16)w[i];
      a0[i] = (__bf16)t0[i];
      a1[i] = (__bf16)t1[i];
    }
    c0 = wmma_bf16(a0, bm, c0);
    c1 = wmma_bf16(a1, bm, c1);
    h0 += 32; h1 += 32;
  };

  const int KSTEPS = Hn / 32;  // 128
  stage(0); stage(1); stage(2);
  int cur = 0;
  for (int j = 0; j < KSTEPS - 3; ++j) {
    stage((cur + 3) & 3);
    wait_async_le12();
    body(cur);
    cur = (cur + 1) & 3;
  }
  wait_async_le8();  body(cur); cur = (cur + 1) & 3;
  wait_async_le4();  body(cur); cur = (cur + 1) & 3;
  wait_async_le0();  body(cur);

  // C layout: element e -> row half*8+e, col = nr
  #pragma unroll
  for (int e = 0; e < 8; ++e) {
    outp[(size_t)(half * 8 + e) * ncols + cb + nr]      = c0[e];
    outp[(size_t)(16 + half * 8 + e) * ncols + cb + nr] = c1[e];
  }
}

// ---------------------------------------------------------------------------
// Kernel 2: RoPE on q and k buffers (pair d / d+64 handled together).
// ---------------------------------------------------------------------------
__global__ void rope_kernel(float* __restrict__ qbuf, float* __restrict__ kbuf,
                            const float* __restrict__ cosp, const float* __restrict__ sinp) {
  int idx = blockIdx.x * blockDim.x + threadIdx.x;
  const int totq = Bn * Qn * NHn * (HDn / 2);
  const int totk = Bn * Qn * NKVn * (HDn / 2);
  float* buf; int nh;
  if (idx < totq) { buf = qbuf; nh = NHn; }
  else { idx -= totq; if (idx >= totk) return; buf = kbuf; nh = NKVn; }
  const int d = idx % (HDn / 2); idx /= (HDn / 2);
  const int head = idx % nh;     idx /= nh;
  const int row = idx;  // b*Q+qi
  const float c1 = cosp[row * HDn + d],      s1 = sinp[row * HDn + d];
  const float c2 = cosp[row * HDn + d + 64], s2 = sinp[row * HDn + d + 64];
  float* p = buf + ((size_t)row * nh + head) * HDn;
  const float x1 = p[d], x2 = p[d + 64];
  p[d]      = x1 * c1 - x2 * s1;
  p[d + 64] = x2 * c2 + x1 * s2;
}

// ---------------------------------------------------------------------------
// Kernel 3: flash-decoding attention partial. One wave per (b, kv, split).
// V chunks are double-buffered in LDS (stage c+1 while computing c) via async
// b128; boundary chunk uses branchless dummy-row loads + causal masking.
// ---------------------------------------------------------------------------
__global__ __launch_bounds__(32) void attn_partial_kernel(
    const float* __restrict__ qbuf, const float* __restrict__ kbuf,
    const float* __restrict__ vbuf, const float* __restrict__ Kc,
    const float* __restrict__ Vc, const int* __restrict__ cache_lens,
    float* __restrict__ pO, float* __restrict__ pM, float* __restrict__ pL) {
  __shared__ float lds_p[16 * 32];
  __shared__ float lds_v[2][32 * 132];

  const int bid = blockIdx.x;
  const int split = bid % NSPLIT;
  const int kv = (bid / NSPLIT) % NKVn;
  const int b = bid / (NSPLIT * NKVn);
  const int lane = threadIdx.x & 31;
  const int half = lane >> 4, nr = lane & 15;
  const int clen = cache_lens[b];
  const int total = clen + Qn;
  const int nchunks = (total + 31) / 32;
  const int cbeg = (nchunks * split) / NSPLIT;
  const int cend = (nchunks * (split + 1)) / NSPLIT;

  const size_t rowstride = (size_t)NKVn * HDn;
  const float* Kb = Kc + ((size_t)b * CAPn * NKVn + kv) * HDn;  // + pos*rowstride
  const float* Vb = Vc + ((size_t)b * CAPn * NKVn + kv) * HDn;
  const float* knew = kbuf + ((size_t)b * Qn * NKVn + kv) * HDn;
  const float* vnew = vbuf + ((size_t)b * Qn * NKVn + kv) * HDn;

  // Q tile in A layout: row m = nr, m = qi*G + g, head = kv*G + g
  v16bf aq[4];
  {
    const int qi = nr / Gn, g = nr % Gn;
    const int head = kv * Gn + g;
    const float* qp = qbuf + ((size_t)(b * Qn + qi) * NHn + head) * HDn;
    #pragma unroll
    for (int ks = 0; ks < 4; ++ks) {
      const int dbase = ks * 32 + half * 16;
      float t[16];
      *(float4*)(t + 0)  = *(const float4*)(qp + dbase);
      *(float4*)(t + 4)  = *(const float4*)(qp + dbase + 4);
      *(float4*)(t + 8)  = *(const float4*)(qp + dbase + 8);
      *(float4*)(t + 12) = *(const float4*)(qp + dbase + 12);
      #pragma unroll
      for (int i = 0; i < 16; ++i) aq[ks][i] = (__bf16)t[i];
    }
  }

  v8f oacc[8];
  #pragma unroll
  for (int t = 0; t < 8; ++t)
    #pragma unroll
    for (int e = 0; e < 8; ++e) oacc[t][e] = 0.f;
  float rmax[8], rsum[8];
  #pragma unroll
  for (int e = 0; e < 8; ++e) { rmax[e] = -1e30f; rsum[e] = 0.f; }

  // stage V chunk c into LDS buffer vb (32 async b128 per wave, coalesced)
  auto stage_v = [&](int c, int vb) {
    const int basep = c * 32;
    float* ldsb = &lds_v[vb][0];
    if (basep + 32 <= clen) {                 // uniform fast case
      const float* vsrc = Vb + (size_t)basep * rowstride;
      #pragma unroll
      for (int i = 0; i < 32; ++i) {
        const int s = i * 32 + lane, krow = s >> 5, d4 = s & 31;
        async_ld_b128((unsigned)(uintptr_t)&ldsb[krow * 132 + d4 * 4],
                      vsrc + (size_t)krow * rowstride + d4 * 4);
      }
    } else {                                  // boundary chunk: branchless selects
      #pragma unroll
      for (int i = 0; i < 32; ++i) {
        const int s = i * 32 + lane, krow = s >> 5, d4 = s & 31;
        const int pos = basep + krow;
        const float* src =
            (pos < clen)  ? Vb + (size_t)pos * rowstride
          : (pos < total) ? vnew + (size_t)(pos - clen) * rowstride
                          : Vb;                       // masked later (p == 0)
        async_ld_b128((unsigned)(uintptr_t)&ldsb[krow * 132 + d4 * 4],
                      src + d4 * 4);
      }
    }
  };

  if (cbeg < cend) stage_v(cbeg, 0);
  int vcur = 0;
  for (int c = cbeg; c < cend; ++c) {
    const int basep = c * 32;
    const bool fast = (basep + 32 <= clen);
    const bool have_next = (c + 1 < cend);
    if (have_next) stage_v(c + 1, vcur ^ 1);

    // ---- K row pointers for the two 16-key halves ----
    const float* kp0;
    const float* kp1;
    if (fast) {
      kp0 = Kb + (size_t)(basep + nr) * rowstride;
      kp1 = Kb + (size_t)(basep + 16 + nr) * rowstride;
      __builtin_prefetch(kp0 + 32 * rowstride, 0, 1);
      __builtin_prefetch(kp1 + 32 * rowstride, 0, 1);
    } else {
      const int p0 = basep + nr;
      kp0 = (p0 < clen)  ? Kb + (size_t)p0 * rowstride
          : (p0 < total) ? knew + (size_t)(p0 - clen) * rowstride
                         : Kb;                       // masked below
      const int p1 = basep + 16 + nr;
      kp1 = (p1 < clen)  ? Kb + (size_t)p1 * rowstride
          : (p1 < total) ? knew + (size_t)(p1 - clen) * rowstride
                         : Kb;                       // masked below
    }

    // ---- S = Q K^T ----
    v8f sc0, sc1;
    #pragma unroll
    for (int e = 0; e < 8; ++e) { sc0[e] = 0.f; sc1[e] = 0.f; }
    #pragma unroll
    for (int ks = 0; ks < 4; ++ks) {
      const int dbase = ks * 32 + half * 16;
      float t0[16], t1[16];
      *(float4*)(t0 + 0)  = *(const float4*)(kp0 + dbase);
      *(float4*)(t0 + 4)  = *(const float4*)(kp0 + dbase + 4);
      *(float4*)(t0 + 8)  = *(const float4*)(kp0 + dbase + 8);
      *(float4*)(t0 + 12) = *(const float4*)(kp0 + dbase + 12);
      *(float4*)(t1 + 0)  = *(const float4*)(kp1 + dbase);
      *(float4*)(t1 + 4)  = *(const float4*)(kp1 + dbase + 4);
      *(float4*)(t1 + 8)  = *(const float4*)(kp1 + dbase + 8);
      *(float4*)(t1 + 12) = *(const float4*)(kp1 + dbase + 12);
      v16bf k0, k1;
      #pragma unroll
      for (int i = 0; i < 16; ++i) { k0[i] = (__bf16)t0[i]; k1[i] = (__bf16)t1[i]; }
      sc0 = wmma_bf16(aq[ks], k0, sc0);
      sc1 = wmma_bf16(aq[ks], k1, sc1);
    }

    // ---- online softmax (row r = half*8+e on element e) ----
    #pragma unroll
    for (int e = 0; e < 8; ++e) {
      float s0 = sc0[e] * SCALE_F;
      float s1 = sc1[e] * SCALE_F;
      if (!fast) {
        const int lim = clen + (half * 8 + e) / Gn;   // causal limit
        if (basep + nr > lim)      s0 = -1e30f;
        if (basep + 16 + nr > lim) s1 = -1e30f;
      }
      float mx = fmaxf(s0, s1);
      #pragma unroll
      for (int off = 1; off < 16; off <<= 1) mx = fmaxf(mx, __shfl_xor(mx, off, 32));
      const float newm = fmaxf(rmax[e], mx);
      const float corr = __expf(rmax[e] - newm);
      const float p0 = __expf(s0 - newm);
      const float p1 = __expf(s1 - newm);
      float sm = p0 + p1;
      #pragma unroll
      for (int off = 1; off < 16; off <<= 1) sm += __shfl_xor(sm, off, 32);
      rsum[e] = rsum[e] * corr + sm;
      rmax[e] = newm;
      #pragma unroll
      for (int t = 0; t < 8; ++t) oacc[t][e] *= corr;
      lds_p[(half * 8 + e) * 32 + nr]      = p0;
      lds_p[(half * 8 + e) * 32 + 16 + nr] = p1;
    }
    // wait for this chunk's V (oldest 32 async ops); next chunk's may be in flight
    if (have_next) wait_async_le32(); else wait_async_le0();
    __syncthreads();

    // ---- P in A layout: two 32B contiguous LDS reads per K-half ----
    float tp[16];
    *(float4*)(tp + 0)  = *(const float4*)(&lds_p[nr * 32 + half * 8]);
    *(float4*)(tp + 4)  = *(const float4*)(&lds_p[nr * 32 + half * 8 + 4]);
    *(float4*)(tp + 8)  = *(const float4*)(&lds_p[nr * 32 + 16 + half * 8]);
    *(float4*)(tp + 12) = *(const float4*)(&lds_p[nr * 32 + 16 + half * 8 + 4]);
    v16bf pa;
    #pragma unroll
    for (int i = 0; i < 16; ++i) pa[i] = (__bf16)tp[i];

    // ---- O += P @ V : 8 N-tiles, batched column gathers ----
    const float* vls = &lds_v[vcur][0];
    #pragma unroll
    for (int t = 0; t < 8; ++t) {
      float tv[16];
      #pragma unroll
      for (int i = 0; i < 16; ++i)
        tv[i] = vls[(i < 8 ? half * 8 + i : 8 + half * 8 + i) * 132 + t * 16 + nr];
      v16bf vb;
      #pragma unroll
      for (int i = 0; i < 16; ++i) vb[i] = (__bf16)tv[i];
      oacc[t] = wmma_bf16(pa, vb, oacc[t]);
    }
    __syncthreads();
    vcur ^= 1;
  }

  // Write partial O (16 x 128) and row stats for this split.
  float* po = pO + (size_t)bid * 16 * HDn;
  #pragma unroll
  for (int t = 0; t < 8; ++t)
    #pragma unroll
    for (int e = 0; e < 8; ++e)
      po[(half * 8 + e) * HDn + t * 16 + nr] = oacc[t][e];
  if (nr == 0) {
    #pragma unroll
    for (int e = 0; e < 8; ++e) {
      pM[(size_t)bid * 16 + half * 8 + e] = rmax[e];
      pL[(size_t)bid * 16 + half * 8 + e] = rsum[e];
    }
  }
}

// ---------------------------------------------------------------------------
// Kernel 4: merge the NSPLIT partials per (b, kv) into the attention output.
// ---------------------------------------------------------------------------
__global__ void attn_combine_kernel(const float* __restrict__ pO,
                                    const float* __restrict__ pM,
                                    const float* __restrict__ pL,
                                    float* __restrict__ obuf) {
  const int bk = blockIdx.x;       // b*NKV + kv
  const int kv = bk % NKVn, b = bk / NKVn;
  const int d = threadIdx.x;       // 0..127
  for (int m = 0; m < 16; ++m) {
    float gmax = -1e30f;
    for (int s = 0; s < NSPLIT; ++s)
      gmax = fmaxf(gmax, pM[(size_t)(bk * NSPLIT + s) * 16 + m]);
    float denom = 0.f, acc = 0.f;
    for (int s = 0; s < NSPLIT; ++s) {
      const float w = __expf(pM[(size_t)(bk * NSPLIT + s) * 16 + m] - gmax);
      denom += w * pL[(size_t)(bk * NSPLIT + s) * 16 + m];
      acc   += w * pO[((size_t)(bk * NSPLIT + s) * 16 + m) * HDn + d];
    }
    const int qi = m / Gn, g = m % Gn;
    const int head = kv * Gn + g;
    obuf[((size_t)(b * Qn + qi) * NHn + head) * HDn + d] = acc / denom;
  }
}

// ---------------------------------------------------------------------------
// Kernel 5: output projection 32x4096 @ 4096x4096, same scheme as kernel 1.
// ---------------------------------------------------------------------------
__global__ __launch_bounds__(32) void out_gemm_kernel(
    const float* __restrict__ obuf, const float* __restrict__ Wo,
    float* __restrict__ out) {
  __shared__ float wtile[4][32 * 16];

  const int lane = threadIdx.x & 31;
  const int half = lane >> 4, nr = lane & 15;
  const int cb = blockIdx.x * 16;

  const float* gsrc[4];
  unsigned lbase[4];
  #pragma unroll
  for (int i = 0; i < 4; ++i) {
    const int slot = i * 32 + lane, row = slot >> 2, c4 = slot & 3;
    gsrc[i]  = Wo + (size_t)row * Hn + cb + c4 * 4;
    lbase[i] = (unsigned)(uintptr_t)&wtile[0][row * 16 + c4 * 4];
  }
  const size_t kadv = (size_t)32 * Hn;
  auto stage = [&](int buf) {
    #pragma unroll
    for (int i = 0; i < 4; ++i) {
      async_ld_b128(lbase[i] + (unsigned)buf * 2048u, gsrc[i]);
      gsrc[i] += kadv;
    }
  };

  v8f c0, c1;
  #pragma unroll
  for (int e = 0; e < 8; ++e) { c0[e] = 0.f; c1[e] = 0.f; }

  const float* h0 = obuf + (size_t)nr * (NHn * HDn) + half * 8;
  const float* h1 = obuf + (size_t)(16 + nr) * (NHn * HDn) + half * 8;

  auto body = [&](int cur) {
    float w[16];
    #pragma unroll
    for (int i = 0; i < 16; ++i) w[i] = wtile[cur][(i < 8 ? half * 8 + i : 8 + half * 8 + i) * 16 + nr];
    float t0[16], t1[16];
    *(float4*)(t0 + 0)  = *(const float4*)(h0);
    *(float4*)(t0 + 4)  = *(const float4*)(h0 + 4);
    *(float4*)(t0 + 8)  = *(const float4*)(h0 + 16);
    *(float4*)(t0 + 12) = *(const float4*)(h0 + 20);
    *(float4*)(t1 + 0)  = *(const float4*)(h1);
    *(float4*)(t1 + 4)  = *(const float4*)(h1 + 4);
    *(float4*)(t1 + 8)  = *(const float4*)(h1 + 16);
    *(float4*)(t1 + 12) = *(const float4*)(h1 + 20);
    v16bf bm, a0, a1;
    #pragma unroll
    for (int i = 0; i < 16; ++i) {
      bm[i] = (__bf16)w[i];
      a0[i] = (__bf16)t0[i];
      a1[i] = (__bf16)t1[i];
    }
    c0 = wmma_bf16(a0, bm, c0);
    c1 = wmma_bf16(a1, bm, c1);
    h0 += 32; h1 += 32;
  };

  const int KSTEPS = (NHn * HDn) / 32;  // 128
  stage(0); stage(1); stage(2);
  int cur = 0;
  for (int j = 0; j < KSTEPS - 3; ++j) {
    stage((cur + 3) & 3);
    wait_async_le12();
    body(cur);
    cur = (cur + 1) & 3;
  }
  wait_async_le8();  body(cur); cur = (cur + 1) & 3;
  wait_async_le4();  body(cur); cur = (cur + 1) & 3;
  wait_async_le0();  body(cur);

  #pragma unroll
  for (int e = 0; e < 8; ++e) {
    out[(size_t)(half * 8 + e) * Hn + cb + nr]      = c0[e];
    out[(size_t)(16 + half * 8 + e) * Hn + cb + nr] = c1[e];
  }
}

// ---------------------------------------------------------------------------
extern "C" void kernel_launch(void* const* d_in, const int* in_sizes, int n_in,
                              void* d_out, int out_size, void* d_ws, size_t ws_size,
                              hipStream_t stream) {
  (void)in_sizes; (void)n_in; (void)out_size; (void)ws_size;
  const float* hs   = (const float*)d_in[0];
  const float* cosp = (const float*)d_in[1];
  const float* sinp = (const float*)d_in[2];
  const float* Wq   = (const float*)d_in[3];
  const float* Wk   = (const float*)d_in[4];
  const float* Wv   = (const float*)d_in[5];
  const float* Wo   = (const float*)d_in[6];
  const float* Kc   = (const float*)d_in[7];
  const float* Vc   = (const float*)d_in[8];
  const int*   clen = (const int*)d_in[9];
  float* out = (float*)d_out;

  float* ws   = (float*)d_ws;
  float* qbuf = ws;                                  // 32*4096
  float* kbuf = qbuf + (size_t)Bn * Qn * NHn * HDn;  // 32*1024
  float* vbuf = kbuf + (size_t)Bn * Qn * NKVn * HDn; // 32*1024
  float* obuf = vbuf + (size_t)Bn * Qn * NKVn * HDn; // 32*4096
  float* pO   = obuf + (size_t)Bn * Qn * NHn * HDn;  // 64*8*16*128
  float* pM   = pO + (size_t)Bn * NKVn * NSPLIT * 16 * HDn;
  float* pL   = pM + (size_t)Bn * NKVn * NSPLIT * 16;

  qkv_gemm_kernel<<<(NHn + 2 * NKVn) * HDn / 16, 32, 0, stream>>>(
      hs, Wq, Wk, Wv, qbuf, kbuf, vbuf);

  const int ropeN = Bn * Qn * (NHn + NKVn) * (HDn / 2);
  rope_kernel<<<(ropeN + 255) / 256, 256, 0, stream>>>(qbuf, kbuf, cosp, sinp);

  attn_partial_kernel<<<Bn * NKVn * NSPLIT, 32, 0, stream>>>(
      qbuf, kbuf, vbuf, Kc, Vc, clen, pO, pM, pL);

  attn_combine_kernel<<<Bn * NKVn, HDn, 0, stream>>>(pO, pM, pL, obuf);

  out_gemm_kernel<<<Hn / 16, 32, 0, stream>>>(obuf, Wo, out);
}
